// SentenceEncoder_24532853195260
// MI455X (gfx1250) — compile-verified, matrix operand order checked by reading, not run on hardware
//
#include <hip/hip_runtime.h>
#include <math.h>

// ---- problem constants (BERT-base, B=8,S=512,D=768,H=12,FF=3072) ----
#define TB   8
#define TS   512
#define TD   768
#define THD  12
#define TDH  64
#define TFF  3072
#define TL   12
#define TM   (TB*TS)   // 4096 rows of activations

typedef _Float16 half_t;
typedef __attribute__((ext_vector_type(16))) _Float16 v16h;
typedef __attribute__((ext_vector_type(8)))  _Float16 v8h;
typedef __attribute__((ext_vector_type(8)))  float    v8f;

// ---------------------------------------------------------------------
// block-wide reductions (256 threads)
// ---------------------------------------------------------------------
__device__ __forceinline__ float block_reduce_sum(float v, float* sbuf) {
    int t = threadIdx.x;
    sbuf[t] = v; __syncthreads();
    for (int s = 128; s > 0; s >>= 1) {
        if (t < s) sbuf[t] += sbuf[t + s];
        __syncthreads();
    }
    float r = sbuf[0]; __syncthreads();
    return r;
}
__device__ __forceinline__ float block_reduce_max(float v, float* sbuf) {
    int t = threadIdx.x;
    sbuf[t] = v; __syncthreads();
    for (int s = 128; s > 0; s >>= 1) {
        if (t < s) sbuf[t] = fmaxf(sbuf[t], sbuf[t + s]);
        __syncthreads();
    }
    float r = sbuf[0]; __syncthreads();
    return r;
}

__device__ __forceinline__ float gelu_exact(float v) {
    return 0.5f * v * (1.0f + erff(v * 0.70710678118654752f));
}

// ---------------------------------------------------------------------
// WMMA fragment load: 16x32 tile slice per ISA layout (16-bit A/B frag)
//   lane 0-15 : row = base+lane,    elems 0..7 -> K=k0..k0+7,    8..15 -> K=k0+16..+23
//   lane 16-31: row = base+lane-16, elems 0..7 -> K=k0+8..k0+15, 8..15 -> K=k0+24..+31
// ---------------------------------------------------------------------
__device__ __forceinline__ v16h load_frag(const half_t* __restrict__ base,
                                          int ld, int rowBase, int k0, int lane) {
    int r  = rowBase + (lane & 15);
    int ks = (lane & 16) ? 8 : 0;
    const half_t* p = base + (size_t)r * ld + k0 + ks;
    v8h lo = *(const v8h*)(p);
    v8h hi = *(const v8h*)(p + 16);
    return __builtin_shufflevector(lo, hi, 0,1,2,3,4,5,6,7,8,9,10,11,12,13,14,15);
}

// ---------------------------------------------------------------------
// Generic WMMA GEMM: C = A(f16 MxK) * Bt(f16 NxK)^T + bias
// 256 threads = 8 waves, waves 4(M) x 2(N); wave tile 32x32 (2x2 WMMA),
// block macro-tile 128x64. grid = (N/64, M/128, batches).
// modes: 0 = f32 row-major [M,N]          (Cf, batch stride strideC)
//        1 = f16 Q/K layout [B,H,S,DH]    (row in [0,B*S), col in [0,D))
//        2 = f16 V^T layout [B,H,DH,S]
//        3 = f16 ctx merge [B,S,D]; batched z=b*H+h, row=s, col=dh
//        4 = f16 row-major [M,N] with exact GELU fused
// ---------------------------------------------------------------------
__global__ __launch_bounds__(256)
void gemm_wmma(const half_t* __restrict__ A, const half_t* __restrict__ Bt,
               const float* __restrict__ bias,
               float* __restrict__ Cf, half_t* __restrict__ Ch,
               int N, int K,
               long long strideA, long long strideB, long long strideC,
               int mode)
{
    int lane  = threadIdx.x & 31;
    int w     = threadIdx.x >> 5;
    int waveM = w >> 1;          // 0..3
    int waveN = w & 1;           // 0..1
    int rowBase = blockIdx.y * 128 + waveM * 32;
    int colBase = blockIdx.x * 64  + waveN * 32;
    int z = blockIdx.z;

    const half_t* Az = A  + (size_t)z * strideA;
    const half_t* Bz = Bt + (size_t)z * strideB;

    v8f acc[2][2] = {};
    for (int k0 = 0; k0 < K; k0 += 32) {
        v16h a0 = load_frag(Az, K, rowBase,      k0, lane);
        v16h a1 = load_frag(Az, K, rowBase + 16, k0, lane);
        v16h b0 = load_frag(Bz, K, colBase,      k0, lane);
        v16h b1 = load_frag(Bz, K, colBase + 16, k0, lane);
        acc[0][0] = __builtin_amdgcn_wmma_f32_16x16x32_f16(false, a0, false, b0,
                                                           (short)0, acc[0][0], false, false);
        acc[0][1] = __builtin_amdgcn_wmma_f32_16x16x32_f16(false, a0, false, b1,
                                                           (short)0, acc[0][1], false, false);
        acc[1][0] = __builtin_amdgcn_wmma_f32_16x16x32_f16(false, a1, false, b0,
                                                           (short)0, acc[1][0], false, false);
        acc[1][1] = __builtin_amdgcn_wmma_f32_16x16x32_f16(false, a1, false, b1,
                                                           (short)0, acc[1][1], false, false);
    }

    int colL   = lane & 15;
    int rowOff = (lane & 16) ? 8 : 0;
    float biasv[2];
    biasv[0] = bias ? bias[colBase + colL]      : 0.0f;
    biasv[1] = bias ? bias[colBase + 16 + colL] : 0.0f;

#pragma unroll
    for (int tm = 0; tm < 2; ++tm) {
#pragma unroll
        for (int tn = 0; tn < 2; ++tn) {
#pragma unroll
            for (int r = 0; r < 8; ++r) {
                int row = rowBase + tm * 16 + rowOff + r;
                int c   = colBase + tn * 16 + colL;
                float v = acc[tm][tn][r] + biasv[tn];
                if (mode == 0) {
                    Cf[(size_t)z * strideC + (size_t)row * N + c] = v;
                } else if (mode == 1) {        // [B,H,S,DH]
                    int b = row >> 9, s = row & (TS - 1);
                    int hh = c >> 6, d = c & 63;
                    Ch[(((size_t)b * THD + hh) * TS + s) * TDH + d] = (half_t)v;
                } else if (mode == 2) {        // [B,H,DH,S]
                    int b = row >> 9, s = row & (TS - 1);
                    int hh = c >> 6, d = c & 63;
                    Ch[(((size_t)b * THD + hh) * TDH + d) * TS + s] = (half_t)v;
                } else if (mode == 3) {        // ctx merge [B,S,D]; z=b*H+h
                    int b = z / THD, hh = z % THD;
                    Ch[((size_t)b * TS + row) * TD + hh * TDH + c] = (half_t)v;
                } else {                        // mode 4: f16 [M,N] + GELU
                    Ch[(size_t)row * N + c] = (half_t)gelu_exact(v);
                }
            }
        }
    }
}

// ---------------------------------------------------------------------
// Weight convert + transpose, LDS-tiled (coalesced both ways):
//   W (f32, KxN row-major) -> Wt (f16, NxK row-major)
// block = 256 thr (32x8), tile 32x32, grid = (N/32, K/32)
// ---------------------------------------------------------------------
__global__ __launch_bounds__(256)
void transpose_w16(const float* __restrict__ W, half_t* __restrict__ Wt,
                   int K, int N)
{
    __shared__ float tile[32][33];
    int n0 = blockIdx.x * 32;
    int k0 = blockIdx.y * 32;
    int tx = threadIdx.x & 31;
    int ty = threadIdx.x >> 5;     // 0..7
#pragma unroll
    for (int j = 0; j < 4; ++j)
        tile[ty + j * 8][tx] = W[(size_t)(k0 + ty + j * 8) * N + n0 + tx];
    __syncthreads();
#pragma unroll
    for (int j = 0; j < 4; ++j)
        Wt[(size_t)(n0 + ty + j * 8) * K + k0 + tx] = (half_t)tile[tx][ty + j * 8];
}

// ---------------------------------------------------------------------
// Embeddings + LayerNorm.  block = row (b*S+s), 256 thr, 3 elems each.
// ---------------------------------------------------------------------
__global__ __launch_bounds__(256)
void embed_ln(const int* __restrict__ ids, const float* __restrict__ wemb,
              const float* __restrict__ pemb, const float* __restrict__ temb,
              const float* __restrict__ g, const float* __restrict__ bb,
              float* __restrict__ hout, half_t* __restrict__ h16)
{
    __shared__ float sbuf[256];
    int row = blockIdx.x;
    int s = row & (TS - 1);
    int tok = ids[row];
    float v[3]; float sum = 0.0f;
#pragma unroll
    for (int j = 0; j < 3; ++j) {
        int d = threadIdx.x + j * 256;
        v[j] = wemb[(size_t)tok * TD + d] + pemb[(size_t)s * TD + d] + temb[d];
        sum += v[j];
    }
    float mu = block_reduce_sum(sum, sbuf) * (1.0f / TD);
    float var = 0.0f;
#pragma unroll
    for (int j = 0; j < 3; ++j) { v[j] -= mu; var += v[j] * v[j]; }
    var = block_reduce_sum(var, sbuf) * (1.0f / TD);
    float rstd = rsqrtf(var + 1e-12f);
#pragma unroll
    for (int j = 0; j < 3; ++j) {
        int d = threadIdx.x + j * 256;
        float o = v[j] * rstd * g[d] + bb[d];
        hout[(size_t)row * TD + d] = o;
        h16 [(size_t)row * TD + d] = (half_t)o;
    }
}

// ---------------------------------------------------------------------
// (pre + residual) -> LayerNorm -> f32 out + f16 out
// ---------------------------------------------------------------------
__global__ __launch_bounds__(256)
void ln_residual(const float* __restrict__ pre, const float* __restrict__ resd,
                 const float* __restrict__ g, const float* __restrict__ bb,
                 float* __restrict__ outf, half_t* __restrict__ outh)
{
    __shared__ float sbuf[256];
    int row = blockIdx.x;
    float v[3]; float sum = 0.0f;
#pragma unroll
    for (int j = 0; j < 3; ++j) {
        size_t idx = (size_t)row * TD + threadIdx.x + j * 256;
        v[j] = pre[idx] + resd[idx];
        sum += v[j];
    }
    float mu = block_reduce_sum(sum, sbuf) * (1.0f / TD);
    float var = 0.0f;
#pragma unroll
    for (int j = 0; j < 3; ++j) { v[j] -= mu; var += v[j] * v[j]; }
    var = block_reduce_sum(var, sbuf) * (1.0f / TD);
    float rstd = rsqrtf(var + 1e-12f);
#pragma unroll
    for (int j = 0; j < 3; ++j) {
        int d = threadIdx.x + j * 256;
        float o = v[j] * rstd * g[d] + bb[d];
        outf[(size_t)row * TD + d] = o;
        outh[(size_t)row * TD + d] = (half_t)o;
    }
}

// ---------------------------------------------------------------------
// scale + softmax + mask + renormalize. block = one score row (512 cols)
// ---------------------------------------------------------------------
__global__ __launch_bounds__(256)
void softmax_mask(const float* __restrict__ scores, const int* __restrict__ mask,
                  half_t* __restrict__ probs)
{
    __shared__ float sbuf[256];
    size_t row = blockIdx.x;                 // (b*H + h)*S + s
    int b = (int)(row / ((size_t)THD * TS));
    const float* src = scores + row * TS;
    int t = threadIdx.x;
    float x0 = src[t]       * 0.125f;        // 1/sqrt(64)
    float x1 = src[t + 256] * 0.125f;
    float mx = block_reduce_max(fmaxf(x0, x1), sbuf);
    float e0 = __expf(x0 - mx);
    float e1 = __expf(x1 - mx);
    float m0 = (float)mask[(size_t)b * TS + t];
    float m1 = (float)mask[(size_t)b * TS + t + 256];
    float s1 = block_reduce_sum(e0 + e1, sbuf);
    float s2 = block_reduce_sum(e0 * m0 + e1 * m1, sbuf);
    float denom = fmaxf(s2, 1e-9f * s1);
    float inv = 1.0f / denom;
    probs[row * TS + t]       = (half_t)(e0 * m0 * inv);
    probs[row * TS + t + 256] = (half_t)(e1 * m1 * inv);
}

// ---------------------------------------------------------------------
// masked mean pool over S + L2 normalize. block = batch b.
// ---------------------------------------------------------------------
__global__ __launch_bounds__(256)
void pool_norm(const float* __restrict__ h, const int* __restrict__ mask,
               float* __restrict__ out)
{
    __shared__ float sbuf[256];
    int b = blockIdx.x;
    float acc[3] = {0.0f, 0.0f, 0.0f};
    float msum = 0.0f;
    for (int s = 0; s < TS; ++s) {
        float mv = (float)mask[(size_t)b * TS + s];
        msum += mv;
        const float* rowp = h + ((size_t)b * TS + s) * TD;
#pragma unroll
        for (int j = 0; j < 3; ++j)
            acc[j] += rowp[threadIdx.x + j * 256] * mv;
    }
    float inv = 1.0f / fmaxf(msum, 1e-6f);
    float sq = 0.0f;
#pragma unroll
    for (int j = 0; j < 3; ++j) { acc[j] *= inv; sq += acc[j] * acc[j]; }
    float tot = block_reduce_sum(sq, sbuf);
    float nrm = fmaxf(sqrtf(tot), 1e-12f);
    float rinv = 1.0f / nrm;
#pragma unroll
    for (int j = 0; j < 3; ++j)
        out[(size_t)b * TD + threadIdx.x + j * 256] = acc[j] * rinv;
}

// ---------------------------------------------------------------------
// host orchestration
// ---------------------------------------------------------------------
extern "C" void kernel_launch(void* const* d_in, const int* in_sizes, int n_in,
                              void* d_out, int out_size, void* d_ws, size_t ws_size,
                              hipStream_t stream)
{
    (void)in_sizes; (void)n_in; (void)out_size; (void)ws_size;

    const int*   ids   = (const int*)  d_in[0];
    const int*   amask = (const int*)  d_in[1];
    const float* wemb  = (const float*)d_in[2];
    const float* pemb  = (const float*)d_in[3];
    const float* temb  = (const float*)d_in[4];
    const float* elng  = (const float*)d_in[5];
    const float* elnb  = (const float*)d_in[6];
    const float* Wq = (const float*)d_in[7];  const float* bq = (const float*)d_in[8];
    const float* Wk = (const float*)d_in[9];  const float* bk = (const float*)d_in[10];
    const float* Wv = (const float*)d_in[11]; const float* bv = (const float*)d_in[12];
    const float* Wo = (const float*)d_in[13]; const float* bo = (const float*)d_in[14];
    const float* g1 = (const float*)d_in[15]; const float* b1 = (const float*)d_in[16];
    const float* Wi = (const float*)d_in[17]; const float* bi = (const float*)d_in[18];
    const float* Wf = (const float*)d_in[19]; const float* bf = (const float*)d_in[20];
    const float* g2 = (const float*)d_in[21]; const float* b2 = (const float*)d_in[22];
    float* out = (float*)d_out;

    char* base = (char*)d_ws;
    size_t off = 0;
    auto alloc = [&](size_t bytes) -> char* {
        char* p = base + off;
        off += (bytes + 255) & ~(size_t)255;
        return p;
    };

    const size_t MD  = (size_t)TM * TD;               // 3.15M elems
    const size_t ATT = (size_t)TB * THD * TS * TS;    // 25.2M elems

    float*  h       = (float*) alloc(MD * 4);
    half_t* x16     = (half_t*)alloc(MD * 2);
    float*  resd    = (float*) alloc(MD * 4);
    half_t* ctx16   = (half_t*)alloc(MD * 2);
    half_t* q16     = (half_t*)alloc(MD * 2);
    half_t* k16     = (half_t*)alloc(MD * 2);
    half_t* vt16    = (half_t*)alloc(MD * 2);
    half_t* wt      = (half_t*)alloc((size_t)TD * TFF * 2);
    float*  scores  = (float*) alloc(ATT * 4);
    half_t* probs   = (half_t*)alloc(ATT * 2);
    float*  cproj   = (float*) alloc(MD * 4);
    half_t* inter16 = (half_t*)alloc((size_t)TM * TFF * 2);

    // embeddings + LN
    embed_ln<<<TM, 256, 0, stream>>>(ids, wemb, pemb, temb, elng, elnb, h, x16);

    const dim3 tDD(TD  / 32, TD  / 32);   // transpose DxD weight
    const dim3 tDF(TFF / 32, TD  / 32);   // Wi: K=D,  N=FF
    const dim3 tFD(TD  / 32, TFF / 32);   // Wf: K=FF, N=D
    const dim3 gProj (TD  / 64, TM / 128, 1);
    const dim3 gFF1  (TFF / 64, TM / 128, 1);
    const dim3 gScore(TS  / 64, TS / 128, TB * THD);
    const dim3 gCtx  (TDH / 64, TS / 128, TB * THD);

    for (int l = 0; l < TL; ++l) {
        const float* wq = Wq + (size_t)l * TD * TD;  const float* bq_l = bq + (size_t)l * TD;
        const float* wk = Wk + (size_t)l * TD * TD;  const float* bk_l = bk + (size_t)l * TD;
        const float* wv = Wv + (size_t)l * TD * TD;  const float* bv_l = bv + (size_t)l * TD;
        const float* wo = Wo + (size_t)l * TD * TD;  const float* bo_l = bo + (size_t)l * TD;
        const float* wi = Wi + (size_t)l * TD * TFF; const float* bi_l = bi + (size_t)l * TFF;
        const float* wf = Wf + (size_t)l * TFF * TD; const float* bf_l = bf + (size_t)l * TD;
        const float* g1_l = g1 + (size_t)l * TD; const float* b1_l = b1 + (size_t)l * TD;
        const float* g2_l = g2 + (size_t)l * TD; const float* b2_l = b2 + (size_t)l * TD;

        // Q = h @ Wq + bq  -> f16 [B,H,S,DH]
        transpose_w16<<<tDD, 256, 0, stream>>>(wq, wt, TD, TD);
        gemm_wmma<<<gProj, 256, 0, stream>>>(x16, wt, bq_l, nullptr, q16,
                                             TD, TD, 0, 0, 0, 1);
        // K -> f16 [B,H,S,DH]
        transpose_w16<<<tDD, 256, 0, stream>>>(wk, wt, TD, TD);
        gemm_wmma<<<gProj, 256, 0, stream>>>(x16, wt, bk_l, nullptr, k16,
                                             TD, TD, 0, 0, 0, 1);
        // V -> f16 transposed [B,H,DH,S]
        transpose_w16<<<tDD, 256, 0, stream>>>(wv, wt, TD, TD);
        gemm_wmma<<<gProj, 256, 0, stream>>>(x16, wt, bv_l, nullptr, vt16,
                                             TD, TD, 0, 0, 0, 2);

        // scores = Q @ K^T  (batched over B*H), f32
        gemm_wmma<<<gScore, 256, 0, stream>>>(q16, k16, nullptr, scores, nullptr,
                                              TS, TDH,
                                              (long long)TS * TDH, (long long)TS * TDH,
                                              (long long)TS * TS, 0);
        // softmax (scale 1/8) + mask + renorm -> f16 probs
        softmax_mask<<<TB * THD * TS, 256, 0, stream>>>(scores, amask, probs);

        // ctx = probs @ V  -> merged f16 [B,S,D]
        gemm_wmma<<<gCtx, 256, 0, stream>>>(probs, vt16, nullptr, nullptr, ctx16,
                                            TDH, TS,
                                            (long long)TS * TS, (long long)TDH * TS,
                                            0, 3);

        // attn_pre = ctx @ Wo + bo  (f32)
        transpose_w16<<<tDD, 256, 0, stream>>>(wo, wt, TD, TD);
        gemm_wmma<<<gProj, 256, 0, stream>>>(ctx16, wt, bo_l, cproj, nullptr,
                                             TD, TD, 0, 0, 0, 0);
        // LN(attn_pre + h) -> resd (f32) + x16 (f16)
        ln_residual<<<TM, 256, 0, stream>>>(cproj, h, g1_l, b1_l, resd, x16);

        // inter16 = gelu(attn @ Wi + bi)  (f16, fused epilogue)
        transpose_w16<<<tDF, 256, 0, stream>>>(wi, wt, TD, TFF);
        gemm_wmma<<<gFF1, 256, 0, stream>>>(x16, wt, bi_l, nullptr, inter16,
                                            TFF, TD, 0, 0, 0, 4);

        // ffn_pre = inter @ Wf + bf (f32)
        transpose_w16<<<tFD, 256, 0, stream>>>(wf, wt, TFF, TD);
        gemm_wmma<<<gProj, 256, 0, stream>>>(inter16, wt, bf_l, cproj, nullptr,
                                             TD, TFF, 0, 0, 0, 0);
        // LN(ffn_pre + resd) -> h (f32) + x16 (f16) for next layer
        ln_residual<<<TM, 256, 0, stream>>>(cproj, resd, g2_l, b2_l, h, x16);
    }

    // masked mean pool + L2 normalize -> out [B, D]
    pool_norm<<<TB, 256, 0, stream>>>(h, amask, out);
}
